// GCN_regression_69578470195569
// MI455X (gfx1250) — matrix operand order repaired and checked by reading, not compiled, not run on hardware
//
#include <hip/hip_runtime.h>

// -------- types --------
typedef __bf16 bf16_t;
typedef __attribute__((ext_vector_type(16))) __bf16 v16bf;
typedef __attribute__((ext_vector_type(8)))  __bf16 v8bf;
typedef __attribute__((ext_vector_type(8)))  float  v8f;

#define DFEAT 256   // feature width D == H == 256

// ---------------- small utility kernels ----------------

__global__ void k_zero_f32(float* __restrict__ p, long long n) {
  long long i = (long long)blockIdx.x * blockDim.x + threadIdx.x;
  if (i < n) p[i] = 0.0f;
}

__global__ void k_f32_to_bf16(const float* __restrict__ in, bf16_t* __restrict__ out,
                              long long n) {
  long long i = (long long)blockIdx.x * blockDim.x + threadIdx.x;
  if (i < n) out[i] = (bf16_t)in[i];
}

// W: [K, 256] f32 row-major  ->  Wt: [256, K] bf16 (N-major, K contiguous)
__global__ void k_transpose_w(const float* __restrict__ W, bf16_t* __restrict__ Wt, int K) {
  long long i = (long long)blockIdx.x * blockDim.x + threadIdx.x;
  if (i < (long long)K * DFEAT) {
    int k = (int)(i >> 8);
    int n = (int)(i & 255);
    Wt[(size_t)n * K + k] = (bf16_t)W[i];
  }
}

// ---------------- graph aggregation (memory-bound part) ----------------

__global__ void k_degree(const int* __restrict__ dst, float* __restrict__ deg, int E) {
  int e = blockIdx.x * blockDim.x + threadIdx.x;
  if (e < E) atomicAdd(&deg[dst[e]], 1.0f);
}

// one thread handles 4 features of one edge: gather bf16, scatter-add f32
__global__ void k_scatter(const int* __restrict__ src, const int* __restrict__ dst,
                          const bf16_t* __restrict__ feat, float* __restrict__ aggf,
                          long long E) {
  long long t = (long long)blockIdx.x * blockDim.x + threadIdx.x;
  if (t >= E * 64) return;
  int e = (int)(t >> 6);
  int q = (int)(t & 63) << 2;
  int s = src[e];
  int d = dst[e];
  const bf16_t* p = feat + (size_t)s * DFEAT + q;
  float* o = aggf + (size_t)d * DFEAT + q;
#pragma unroll
  for (int j = 0; j < 4; ++j) atomicAdd(o + j, (float)p[j]);
}

// mean (deg clamped to >=1) and convert to bf16
__global__ void k_mean_bf16(const float* __restrict__ aggf, const float* __restrict__ deg,
                            bf16_t* __restrict__ aggb, long long n) {
  long long i = (long long)blockIdx.x * blockDim.x + threadIdx.x;
  if (i < n) {
    int row = (int)(i >> 8);
    float dv = fmaxf(deg[row], 1.0f);
    aggb[i] = (bf16_t)(aggf[i] * (1.0f / dv));
  }
}

// ---------------- WMMA GEMM: out = relu(concat(A0,A1) @ W + b) ----------------
// A0, A1: [M, 256] bf16 (A1 may be null -> K = 256)
// Wt:     [256, Ktot] bf16 (N-major, pre-transposed)
// bias:   [256] f32,  out: [M, 256] bf16
__global__ void __launch_bounds__(256)
k_sage_gemm(const bf16_t* __restrict__ A0, const bf16_t* __restrict__ A1,
            const bf16_t* __restrict__ Wt, const float* __restrict__ bias,
            bf16_t* __restrict__ out, int M) {
  const int lane = threadIdx.x & 31;
  const int wave = threadIdx.x >> 5;
  const int tile = blockIdx.x * (blockDim.x >> 5) + wave;

  const int tiles_n = DFEAT / 16;                  // 16
  const int tiles_m = (M + 15) >> 4;
  if (tile >= tiles_m * tiles_n) return;           // wave-uniform exit
  const int tm = tile / tiles_n;
  const int tn = tile - tm * tiles_n;
  const int m0 = tm << 4;
  const int n0 = tn << 4;

  const int hf = lane >> 4;                        // half-wave select
  const int lr = lane & 15;
  int mA = m0 + lr; if (mA >= M) mA = M - 1;       // clamp (uniform shapes anyway)
  const int nB = n0 + lr;
  const int Ktot = A1 ? 2 * DFEAT : DFEAT;

  const bf16_t* wrow = Wt + (size_t)nB * Ktot;

  v8f acc = {};

  // ISA 7.12.2 fragment layouts:
  //  A (16x32 bf16): lane=row M; elems 0-7 = K k0+8*hf .. +7, elems 8-15 = +16
  //  B (32x16 bf16): lane=col N; elems 0-7 = K k0+8*hf .. +7, elems 8-15 = +16
  auto mac_panel = [&](const bf16_t* A, int kwoff) {
    const bf16_t* arow = A + (size_t)mA * DFEAT;
#pragma unroll
    for (int k0 = 0; k0 < DFEAT; k0 += 32) {
      union { v16bf v; v8bf h[2]; } a, b;
      a.h[0] = *(const v8bf*)(arow + k0 + hf * 8);
      a.h[1] = *(const v8bf*)(arow + k0 + hf * 8 + 16);
      const bf16_t* wp = wrow + kwoff + k0 + hf * 8;
      b.h[0] = *(const v8bf*)(wp);
      b.h[1] = *(const v8bf*)(wp + 16);
      acc = __builtin_amdgcn_wmma_f32_16x16x32_bf16(false, a.v, false, b.v,
                                                    (short)0, acc, false, false);
    }
  };
  mac_panel(A0, 0);
  if (A1) mac_panel(A1, DFEAT);

  // C/D layout: lane=col N; VGPR r holds row M = m0 + r + 8*hf
  const float bn = bias[nB];
#pragma unroll
  for (int r = 0; r < 8; ++r) {
    int m = m0 + r + hf * 8;
    float v = acc[r] + bn;
    v = v > 0.0f ? v : 0.0f;
    if (m < M) out[(size_t)m * DFEAT + nB] = (bf16_t)v;
  }
}

// ---------------- final head: out[m] = s[m,:] @ Wl2 + bl2 ----------------
__global__ void k_out_dot(const bf16_t* __restrict__ s, const float* __restrict__ Wl2,
                          const float* __restrict__ bl2, float* __restrict__ out, int M) {
  int gw = (int)(((long long)blockIdx.x * blockDim.x + threadIdx.x) >> 5);
  int lane = threadIdx.x & 31;
  if (gw >= M) return;
  const bf16_t* row = s + (size_t)gw * DFEAT + lane * 8;
  const float* w = Wl2 + lane * 8;
  float acc = 0.0f;
#pragma unroll
  for (int j = 0; j < 8; ++j) acc += (float)row[j] * w[j];
#pragma unroll
  for (int off = 16; off > 0; off >>= 1) acc += __shfl_down(acc, off, 32);
  if (lane == 0) out[gw] = acc + bl2[0];
}

// ---------------- launcher ----------------
extern "C" void kernel_launch(void* const* d_in, const int* in_sizes, int n_in,
                              void* d_out, int out_size, void* d_ws, size_t ws_size,
                              hipStream_t stream) {
  (void)n_in; (void)out_size; (void)ws_size;

  const float* x   = (const float*)d_in[0];
  const int*   ei  = (const int*)d_in[1];
  const float* W1  = (const float*)d_in[2];
  const float* b1  = (const float*)d_in[3];
  const float* W2  = (const float*)d_in[4];
  const float* b2  = (const float*)d_in[5];
  const float* Wl1 = (const float*)d_in[6];
  const float* bl1 = (const float*)d_in[7];
  const float* Wl2 = (const float*)d_in[8];
  const float* bl2 = (const float*)d_in[9];
  float* out = (float*)d_out;

  const int N = in_sizes[0] / DFEAT;
  const int E = in_sizes[1] / 2;
  const int* srcI = ei;
  const int* dstI = ei + E;

  // workspace carve-out
  char* ws = (char*)d_ws;
  size_t off = 0;
  auto carve = [&](size_t bytes) -> void* {
    void* p = ws + off;
    off += (bytes + 255) & ~(size_t)255;
    return p;
  };
  const size_t featB = (size_t)N * DFEAT;
  bf16_t* bufA = (bf16_t*)carve(featB * 2);            // activations ping
  bf16_t* bufB = (bf16_t*)carve(featB * 2);            // activations pong
  bf16_t* aggb = (bf16_t*)carve(featB * 2);            // mean-agg bf16
  float*  aggf = (float*) carve(featB * 4);            // scatter accumulator
  float*  deg  = (float*) carve((size_t)N * 4);
  bf16_t* w1t  = (bf16_t*)carve((size_t)2 * DFEAT * DFEAT * 2);
  bf16_t* w2t  = (bf16_t*)carve((size_t)2 * DFEAT * DFEAT * 2);
  bf16_t* wl1t = (bf16_t*)carve((size_t)DFEAT * DFEAT * 2);

  const int T = 256;
  const long long nf = (long long)N * DFEAT;
  const unsigned gF = (unsigned)((nf + T - 1) / T);

  // prep: bf16 activations + transposed bf16 weights + degrees (once)
  k_f32_to_bf16<<<gF, T, 0, stream>>>(x, bufA, nf);
  k_transpose_w<<<(2 * DFEAT * DFEAT + T - 1) / T, T, 0, stream>>>(W1, w1t, 2 * DFEAT);
  k_transpose_w<<<(2 * DFEAT * DFEAT + T - 1) / T, T, 0, stream>>>(W2, w2t, 2 * DFEAT);
  k_transpose_w<<<(DFEAT * DFEAT + T - 1) / T, T, 0, stream>>>(Wl1, wl1t, DFEAT);
  k_zero_f32<<<(N + T - 1) / T, T, 0, stream>>>(deg, N);
  k_degree<<<(E + T - 1) / T, T, 0, stream>>>(dstI, deg, E);

  const long long st = (long long)E * 64;
  const unsigned gS = (unsigned)((st + T - 1) / T);
  const int tiles = ((N + 15) / 16) * (DFEAT / 16);
  const unsigned gG = (unsigned)((tiles + 7) / 8);     // 8 waves / 256-thread block

  // ---- layer 1: h1 = relu([x, mean(x[src]->dst)] @ W1 + b1) ----
  k_zero_f32<<<gF, T, 0, stream>>>(aggf, nf);
  k_scatter<<<gS, T, 0, stream>>>(srcI, dstI, bufA, aggf, (long long)E);
  k_mean_bf16<<<gF, T, 0, stream>>>(aggf, deg, aggb, nf);
  k_sage_gemm<<<gG, 256, 0, stream>>>(bufA, aggb, w1t, b1, bufB, N);

  // ---- layer 2: h2 = relu([h1, mean(h1[src]->dst)] @ W2 + b2) ----
  k_zero_f32<<<gF, T, 0, stream>>>(aggf, nf);
  k_scatter<<<gS, T, 0, stream>>>(srcI, dstI, bufB, aggf, (long long)E);
  k_mean_bf16<<<gF, T, 0, stream>>>(aggf, deg, aggb, nf);
  k_sage_gemm<<<gG, 256, 0, stream>>>(bufB, aggb, w2t, b2, bufA, N);

  // ---- head: s = relu(h2 @ Wl1 + bl1); out = s @ Wl2 + bl2 ----
  k_sage_gemm<<<gG, 256, 0, stream>>>(bufA, (const bf16_t*)nullptr, wl1t, bl1, bufB, N);
  k_out_dot<<<(unsigned)(((long long)N * 32 + T - 1) / T), T, 0, stream>>>(bufB, Wl2, bl2, out, N);
}
